// MambaBlock1D_26474178412653
// MI455X (gfx1250) — matrix operand order repaired and checked by reading, not compiled, not run on hardware
//
#include <hip/hip_runtime.h>
#include <hip/hip_bf16.h>
#include <cstddef>

// ---------------- problem constants ----------------
#define BB    4
#define TT    8192
#define DM    256        // d_model
#define DI    512        // d_inner
#define DXZ   1024       // 2*d_inner
#define RK    16         // dt_rank
#define NS    16         // d_state
#define NCH   16         // scan chunks
#define CHL   (TT / NCH) // 512 steps per chunk

typedef __attribute__((ext_vector_type(16))) __bf16 v16bf;
typedef __attribute__((ext_vector_type(8)))  float  v8f;

static __device__ __forceinline__ float silu_f(float v) {
    return v / (1.f + __expf(-v));
}

// WMMA A-fragment position for element (k, m) in fragment-major LDS layout:
// fragment kf = k/32 holds 32 lanes x 16 contiguous bf16 (32B per lane).
static __device__ __forceinline__ int fragpos(int k, int m) {
    const int kf  = k >> 5;
    const int off = k & 31;
    const int lh  = (off >> 4) & 1;                      // lane-half select
    const int e   = off & 15;                            // element in frag
    return ((kf << 5) + m + (lh << 4)) * 16 + e;
}

static __device__ __forceinline__ v16bf ld_frag(const __bf16* p) {
    return *(const v16bf*)p;
}

// =====================================================================
// K0: pack f32 weights -> bf16 rows (optionally zero-padded to dstride)
// =====================================================================
__global__ void k_pack_bf16(const float* __restrict__ src, __bf16* __restrict__ dst,
                            int rows, int cols, int dstride)
{
    const int idx = blockIdx.x * blockDim.x + threadIdx.x;
    if (idx >= rows * dstride) return;
    const int r = idx / dstride, c = idx % dstride;
    dst[idx] = (c < cols) ? (__bf16)src[r * cols + c] : (__bf16)0.f;
}

// =====================================================================
// K1: fused channel-LayerNorm + in_proj GEMM (M=16 t-rows, N=1024, K=256)
// grid = B*T/16 blocks, 128 threads (4 waves); wave does 4 groups x 4 tiles.
// =====================================================================
__global__ __launch_bounds__(128) void k_ln_inproj(
    const float*  __restrict__ x,      // (B, DM, T)
    const float*  __restrict__ lnw, const float* __restrict__ lnb,
    const __bf16* __restrict__ wbf,    // (DXZ, DM) bf16 row-major
    float* __restrict__ xz)            // (B, T, DXZ)
{
    __shared__ float xs[DM * 16];
    __shared__ __align__(32) __bf16 xnf[(DM / 32) * 32 * 16]; // 8 frags, 8KB
    __shared__ float mu_s[16], rs_s[16];

    const int tid = threadIdx.x;
    const int mt  = blockIdx.x;
    const int b   = mt / (TT / 16);
    const int t0  = (mt % (TT / 16)) * 16;

    for (int idx = tid; idx < DM * 16; idx += 128) {
        const int c = idx >> 4, j = idx & 15;
        xs[idx] = x[((size_t)(b * DM + c)) * TT + t0 + j];
    }
    __syncthreads();

    if (tid < 16) {
        float s = 0.f, s2 = 0.f;
        for (int c = 0; c < DM; ++c) { float v = xs[c * 16 + tid]; s += v; s2 += v * v; }
        const float mu  = s * (1.f / DM);
        const float var = s2 * (1.f / DM) - mu * mu;
        mu_s[tid] = mu;
        rs_s[tid] = rsqrtf(var + 1e-5f);
    }
    __syncthreads();

    for (int idx = tid; idx < DM * 16; idx += 128) {
        const int c = idx >> 4, j = idx & 15;
        const float v = (xs[idx] - mu_s[j]) * rs_s[j] * lnw[c] + lnb[c];
        xnf[fragpos(c, j)] = (__bf16)v;
    }
    __syncthreads();

    const int wave = tid >> 5, lane = tid & 31;
    const int kb2  = (lane >> 4) * 16;   // B-frag K offset
    const int ncol = lane & 15;
    const size_t rowbase = (size_t)b * TT + t0;

    for (int g = 0; g < 4; ++g) {
        v8f acc[4];
#pragma unroll
        for (int q = 0; q < 4; ++q) acc[q] = (v8f){};
        for (int kf = 0; kf < 8; ++kf) {
            const v16bf a = ld_frag(&xnf[((kf << 5) + lane) * 16]);
#pragma unroll
            for (int q = 0; q < 4; ++q) {
                const int n0 = wave * 256 + (g * 4 + q) * 16;
                const v16bf bm = *(const v16bf*)(wbf + (size_t)(n0 + ncol) * DM
                                                     + (kf << 5) + kb2);
                acc[q] = __builtin_amdgcn_wmma_f32_16x16x32_bf16(
                             false, a, false, bm, (short)0, acc[q], false, false);
            }
        }
#pragma unroll
        for (int q = 0; q < 4; ++q) {
            const int n0 = wave * 256 + (g * 4 + q) * 16;
#pragma unroll
            for (int r = 0; r < 8; ++r) {
                const int m = r + ((lane >> 4) << 3);
                xz[(rowbase + m) * DXZ + n0 + ncol] = acc[q][r];
            }
        }
    }
}

// =====================================================================
// K2: causal depthwise conv (4 taps over T) + bias + SiLU  ->  u
// =====================================================================
__global__ void k_conv_silu(const float* __restrict__ xz,
                            const float* __restrict__ cw,
                            const float* __restrict__ cb,
                            float* __restrict__ u)          // (B,T,DI)
{
    const size_t g = (size_t)blockIdx.x * blockDim.x + threadIdx.x;
    if (g >= (size_t)BB * TT * DI) return;
    const int    d  = (int)(g % DI);
    const size_t bt = g / DI;
    const int    t  = (int)(bt % TT);
    const int    b  = (int)(bt / TT);
    float s = cb[d];
#pragma unroll
    for (int k = 0; k < 4; ++k) {
        const int tt = t - 3 + k;
        if (tt >= 0)
            s += cw[d * 4 + k] * xz[(((size_t)b * TT + tt) * DXZ) + d];
    }
    u[g] = silu_f(s);
}

// =====================================================================
// K3: fused x_proj GEMM (16x512 * 512x48) + dt_proj GEMM (16x32pad * 32x512)
//     + softplus.  Writes Bm/Cm to bc, delta to delta.
// =====================================================================
__global__ __launch_bounds__(128) void k_xproj_dtproj(
    const float*  __restrict__ u,      // (B,T,DI)
    const __bf16* __restrict__ xwbf,   // (48, DI) bf16
    const __bf16* __restrict__ dtwbf,  // (DI, 32) bf16, K zero-padded
    const float*  __restrict__ dtbias,
    float* __restrict__ bc,            // (B,T,32)
    float* __restrict__ delta)         // (B,T,DI)
{
    __shared__ __align__(32) __bf16 ub[(DI / 32) * 32 * 16];  // 16 frags, 16KB
    __shared__ __align__(32) __bf16 dts[32 * 16];             // dt frag, 1KB

    const int tid = threadIdx.x;
    const int mt  = blockIdx.x;
    const int b   = mt / (TT / 16);
    const int t0  = (mt % (TT / 16)) * 16;
    const size_t rowbase = (size_t)b * TT + t0;

    for (int idx = tid; idx < 32 * 16; idx += 128) dts[idx] = (__bf16)0.f;
    for (int idx = tid; idx < 16 * DI; idx += 128) {
        const int m = idx / DI, d = idx % DI;
        ub[fragpos(d, m)] = (__bf16)u[(rowbase + m) * DI + d];
    }
    __syncthreads();

    const int wave = tid >> 5, lane = tid & 31;
    const int kb2  = (lane >> 4) * 16;
    const int ncol = lane & 15;

    // ---- GEMM1: waves 0..2 each produce one 16x16 tile of (dt|Bm|Cm) ----
    if (wave < 3) {
        const int n0 = wave * 16;
        v8f acc = {};
        for (int kf = 0; kf < 16; ++kf) {
            const v16bf a  = ld_frag(&ub[((kf << 5) + lane) * 16]);
            const v16bf bm = *(const v16bf*)(xwbf + (size_t)(n0 + ncol) * DI
                                                  + (kf << 5) + kb2);
            acc = __builtin_amdgcn_wmma_f32_16x16x32_bf16(
                      false, a, false, bm, (short)0, acc, false, false);
        }
        if (wave == 0) {           // dt tile -> LDS fragment (k = rank col)
#pragma unroll
            for (int r = 0; r < 8; ++r) {
                const int m = r + ((lane >> 4) << 3);
                dts[fragpos(ncol, m)] = (__bf16)acc[r];
            }
        } else {                   // Bm / Cm -> global
            const int off = (wave == 1) ? 0 : 16;
#pragma unroll
            for (int r = 0; r < 8; ++r) {
                const int m = r + ((lane >> 4) << 3);
                bc[(rowbase + m) * 32 + off + ncol] = acc[r];
            }
        }
    }
    __syncthreads();

    // ---- GEMM2: delta = softplus(dt * dt_proj_w^T + bias), N=512 ----
    const v16bf a = ld_frag(&dts[lane * 16]);
    for (int i = 0; i < 8; ++i) {
        const int n0 = (wave * 8 + i) * 16;
        const v16bf bm = *(const v16bf*)(dtwbf + (size_t)(n0 + ncol) * 32 + kb2);
        v8f acc = {};
        acc = __builtin_amdgcn_wmma_f32_16x16x32_bf16(
                  false, a, false, bm, (short)0, acc, false, false);
#pragma unroll
        for (int r = 0; r < 8; ++r) {
            const int m = r + ((lane >> 4) << 3);
            const float v  = acc[r] + dtbias[n0 + ncol];
            const float sp = (v > 20.f) ? v : log1pf(__expf(v));
            delta[(rowbase + m) * DI + n0 + ncol] = sp;
        }
    }
}

// =====================================================================
// K4: scan pass 1 — per (b,d,chunk): decay product P[n], local final F[n]
// =====================================================================
__global__ void k_scan1(const float* __restrict__ delta,
                        const float* __restrict__ u,
                        const float* __restrict__ bc,
                        const float* __restrict__ A_log,
                        float* __restrict__ P, float* __restrict__ F)
{
    const int g  = blockIdx.x * blockDim.x + threadIdx.x;
    const int d  = g % DI;
    const int ch = (g / DI) % NCH;
    const int b  = g / (DI * NCH);

    float a[NS], p[NS], f[NS];
#pragma unroll
    for (int n = 0; n < NS; ++n) {
        a[n] = -__expf(A_log[d * NS + n]);
        p[n] = 1.f; f[n] = 0.f;
    }
    const int tstart = ch * CHL;
    for (int tt = 0; tt < CHL; ++tt) {
        const size_t row = (size_t)b * TT + tstart + tt;
        if (tt + 8 < CHL) {   // hide 2KB-strided latency -> global_prefetch_b8
            __builtin_prefetch(&delta[(row + 8) * DI + d], 0, 1);
            __builtin_prefetch(&u[(row + 8) * DI + d], 0, 1);
        }
        const float dl = delta[row * DI + d];
        const float du = dl * u[row * DI + d];
        const float* bcr = bc + row * 32;
#pragma unroll
        for (int n = 0; n < NS; ++n) {
            const float dA = __expf(dl * a[n]);
            p[n] *= dA;
            f[n] = dA * f[n] + du * bcr[n];
        }
    }
    const size_t base = (((size_t)b * DI + d) * NCH + ch) * NS;
#pragma unroll
    for (int n = 0; n < NS; ++n) { P[base + n] = p[n]; F[base + n] = f[n]; }
}

// =====================================================================
// K5: combine chunk boundary states
// =====================================================================
__global__ void k_combine(const float* __restrict__ P, const float* __restrict__ F,
                          float* __restrict__ Hin)
{
    const int g = blockIdx.x * blockDim.x + threadIdx.x;
    const int d = g % DI;
    const int b = g / DI;
    float h[NS];
#pragma unroll
    for (int n = 0; n < NS; ++n) h[n] = 0.f;
    for (int ch = 0; ch < NCH; ++ch) {
        const size_t base = (((size_t)b * DI + d) * NCH + ch) * NS;
#pragma unroll
        for (int n = 0; n < NS; ++n) {
            Hin[base + n] = h[n];
            h[n] = P[base + n] * h[n] + F[base + n];
        }
    }
}

// =====================================================================
// K6: scan pass 2 — correct incoming state; y = h*C, +u*D, *silu(z);
//     y written into dead x_in half of xz.
// =====================================================================
__global__ void k_scan2(const float* __restrict__ delta,
                        const float* __restrict__ u,
                        const float* __restrict__ bc,
                        const float* __restrict__ A_log,
                        const float* __restrict__ Hin,
                        const float* __restrict__ Dv,
                        float* __restrict__ xz)
{
    const int g  = blockIdx.x * blockDim.x + threadIdx.x;
    const int d  = g % DI;
    const int ch = (g / DI) % NCH;
    const int b  = g / (DI * NCH);

    float a[NS], h[NS];
    const size_t base = (((size_t)b * DI + d) * NCH + ch) * NS;
#pragma unroll
    for (int n = 0; n < NS; ++n) {
        a[n] = -__expf(A_log[d * NS + n]);
        h[n] = Hin[base + n];
    }
    const float dd = Dv[d];
    const int tstart = ch * CHL;
    for (int tt = 0; tt < CHL; ++tt) {
        const size_t row = (size_t)b * TT + tstart + tt;
        if (tt + 8 < CHL) {
            __builtin_prefetch(&delta[(row + 8) * DI + d], 0, 1);
            __builtin_prefetch(&u[(row + 8) * DI + d], 0, 1);
            __builtin_prefetch(&xz[(row + 8) * DXZ + DI + d], 0, 1);
        }
        const float dl = delta[row * DI + d];
        const float uu = u[row * DI + d];
        const float du = dl * uu;
        const float* bcr = bc + row * 32;
        float y = 0.f;
#pragma unroll
        for (int n = 0; n < NS; ++n) {
            const float dA = __expf(dl * a[n]);
            h[n] = dA * h[n] + du * bcr[n];
            y += h[n] * bcr[16 + n];
        }
        y += uu * dd;
        const float zv = xz[row * DXZ + DI + d];
        xz[row * DXZ + d] = y * silu_f(zv);
    }
}

// =====================================================================
// K7: out_proj GEMM (M=16 t, N=256 c, K=512) + residual add, (B,C,T) out
// =====================================================================
__global__ __launch_bounds__(128) void k_outproj(
    const float*  __restrict__ xz,    // y in cols [0:DI)
    const __bf16* __restrict__ owbf,  // (DM, DI) bf16
    const float*  __restrict__ x,     // residual (B, DM, T)
    float* __restrict__ out)          // (B, DM, T)
{
    __shared__ __align__(32) __bf16 yb[(DI / 32) * 32 * 16]; // 16 frags, 16KB

    const int tid = threadIdx.x;
    const int mt  = blockIdx.x;
    const int b   = mt / (TT / 16);
    const int t0  = (mt % (TT / 16)) * 16;
    const size_t rowbase = (size_t)b * TT + t0;

    for (int idx = tid; idx < 16 * DI; idx += 128) {
        const int m = idx / DI, d = idx % DI;
        yb[fragpos(d, m)] = (__bf16)xz[(rowbase + m) * DXZ + d];
    }
    __syncthreads();

    const int wave = tid >> 5, lane = tid & 31;
    const int kb2  = (lane >> 4) * 16;
    const int ncol = lane & 15;

    v8f acc[4];
#pragma unroll
    for (int q = 0; q < 4; ++q) acc[q] = (v8f){};
    for (int kf = 0; kf < 16; ++kf) {
        const v16bf a = ld_frag(&yb[((kf << 5) + lane) * 16]);
#pragma unroll
        for (int q = 0; q < 4; ++q) {
            const int n0 = (wave * 4 + q) * 16;
            const v16bf bm = *(const v16bf*)(owbf + (size_t)(n0 + ncol) * DI
                                                  + (kf << 5) + kb2);
            acc[q] = __builtin_amdgcn_wmma_f32_16x16x32_bf16(
                         false, a, false, bm, (short)0, acc[q], false, false);
        }
    }
#pragma unroll
    for (int q = 0; q < 4; ++q) {
        const int n0 = (wave * 4 + q) * 16;
#pragma unroll
        for (int r = 0; r < 8; ++r) {
            const int m = r + ((lane >> 4) << 3);
            const int c = n0 + ncol;
            const size_t addr = ((size_t)b * DM + c) * TT + t0 + m;
            out[addr] = acc[q][r] + x[addr];
        }
    }
}

// =====================================================================
extern "C" void kernel_launch(void* const* d_in, const int* in_sizes, int n_in,
                              void* d_out, int out_size, void* d_ws, size_t ws_size,
                              hipStream_t stream) {
    (void)in_sizes; (void)n_in; (void)out_size; (void)ws_size;
    const float* x      = (const float*)d_in[0];
    const float* ln_w   = (const float*)d_in[1];
    const float* ln_b   = (const float*)d_in[2];
    const float* inw    = (const float*)d_in[3];
    const float* convw  = (const float*)d_in[4];
    const float* convb  = (const float*)d_in[5];
    const float* xprojw = (const float*)d_in[6];
    const float* dtw    = (const float*)d_in[7];
    const float* dtb    = (const float*)d_in[8];
    const float* A_log  = (const float*)d_in[9];
    const float* Dvec   = (const float*)d_in[10];
    const float* outw   = (const float*)d_in[11];
    float* out = (float*)d_out;
    float* ws  = (float*)d_ws;

    // ---- workspace layout (float elements) ----
    const size_t XZ    = 0;                                  // (B,T,1024)
    const size_t U     = XZ    + (size_t)BB * TT * DXZ;
    const size_t DELTA = U     + (size_t)BB * TT * DI;
    const size_t BC    = DELTA + (size_t)BB * TT * DI;
    const size_t PP    = BC    + (size_t)BB * TT * 32;
    const size_t FP    = PP    + (size_t)BB * DI * NCH * NS;
    const size_t HIN   = FP    + (size_t)BB * DI * NCH * NS;
    const size_t BFB   = HIN   + (size_t)BB * DI * NCH * NS;

    // ---- bf16 weight region (bf16 elements, 32B-aligned sub-offsets) ----
    __bf16* wsbf = (__bf16*)(ws + BFB);
    const size_t INWBF = 0;
    const size_t XWBF  = INWBF + (size_t)DXZ * DM;   // 262144
    const size_t DTWBF = XWBF  + (size_t)48 * DI;    // +24576
    const size_t OWBF  = DTWBF + (size_t)DI * 32;    // +16384

    // ---- weight pre-pack (f32 -> bf16, dt_proj K-padded to 32) ----
    k_pack_bf16<<<(DXZ * DM) / 256, 256, 0, stream>>>(inw,    wsbf + INWBF, DXZ, DM, DM);
    k_pack_bf16<<<(48 * DI) / 256, 256, 0, stream>>>(xprojw, wsbf + XWBF,  48,  DI, DI);
    k_pack_bf16<<<(DI * 32) / 256, 256, 0, stream>>>(dtw,    wsbf + DTWBF, DI,  RK, 32);
    k_pack_bf16<<<(DM * DI) / 256, 256, 0, stream>>>(outw,   wsbf + OWBF,  DM,  DI, DI);

    const int mblocks = BB * TT / 16;                        // 2048

    k_ln_inproj<<<mblocks, 128, 0, stream>>>(x, ln_w, ln_b, wsbf + INWBF, ws + XZ);

    k_conv_silu<<<(BB * TT * DI) / 256, 256, 0, stream>>>(ws + XZ, convw, convb, ws + U);

    k_xproj_dtproj<<<mblocks, 128, 0, stream>>>(ws + U, wsbf + XWBF, wsbf + DTWBF,
                                                dtb, ws + BC, ws + DELTA);

    k_scan1<<<(BB * NCH * DI) / 256, 256, 0, stream>>>(ws + DELTA, ws + U, ws + BC,
                                                       A_log, ws + PP, ws + FP);

    k_combine<<<(BB * DI) / 256, 256, 0, stream>>>(ws + PP, ws + FP, ws + HIN);

    k_scan2<<<(BB * NCH * DI) / 256, 256, 0, stream>>>(ws + DELTA, ws + U, ws + BC,
                                                       A_log, ws + HIN, Dvec, ws + XZ);

    k_outproj<<<mblocks, 128, 0, stream>>>(ws + XZ, outw ? wsbf + OWBF : wsbf + OWBF,
                                           x, out);
}
